// MultiHeadAttention_17248588661430
// MI455X (gfx1250) — matrix operand order repaired and checked by reading, not compiled, not run on hardware
//
#include <hip/hip_runtime.h>

// Problem constants
#define WB  8      // batch
#define WS  1024   // seq len
#define WD  768    // model dim
#define WH  12     // heads
#define WBH 96     // B*H

typedef __attribute__((ext_vector_type(16))) _Float16 v16h;
typedef __attribute__((ext_vector_type(8)))  _Float16 v8h;
typedef __attribute__((ext_vector_type(4)))  _Float16 v4h;
typedef __attribute__((ext_vector_type(8)))  float    v8f;

union Frag16 { v16h v; v8h h2[2]; _Float16 f[16]; };
union FragC  { v8f  v; float f[8]; };

// ---- WMMA f16 16x16x32, f32 accumulate ----
__device__ __forceinline__ v8f wmma_f16(v16h a, v16h b, v8f c) {
  return __builtin_amdgcn_wmma_f32_16x16x32_f16(false, a, false, b, (short)0, c,
                                                false, false);
}

// A-matrix fragment (16x32): per-lane row, contiguous k in [half*16, +16)
__device__ __forceinline__ v16h load_afrag(const _Float16* row32, int half) {
  Frag16 u;
  u.h2[0] = *(const v8h*)(row32 + half * 16);
  u.h2[1] = *(const v8h*)(row32 + half * 16 + 8);
  return u.v;
}
// B-matrix fragment (32x16): per-lane column, contiguous k in [half*16, +16)
__device__ __forceinline__ v16h load_bfrag(const _Float16* col32, int half) {
  Frag16 u;
  u.h2[0] = *(const v8h*)(col32 + half * 16);
  u.h2[1] = *(const v8h*)(col32 + half * 16 + 8);
  return u.v;
}
// B fragment from an f32 row (vectorized convert on the fly)
__device__ __forceinline__ v16h load_bfrag_f32(const float* col32, int half) {
  const float4* p = (const float4*)(col32 + half * 16);
  float4 a0 = p[0], a1 = p[1], a2 = p[2], a3 = p[3];
  Frag16 u;
  u.f[0]  = (_Float16)a0.x; u.f[1]  = (_Float16)a0.y;
  u.f[2]  = (_Float16)a0.z; u.f[3]  = (_Float16)a0.w;
  u.f[4]  = (_Float16)a1.x; u.f[5]  = (_Float16)a1.y;
  u.f[6]  = (_Float16)a1.z; u.f[7]  = (_Float16)a1.w;
  u.f[8]  = (_Float16)a2.x; u.f[9]  = (_Float16)a2.y;
  u.f[10] = (_Float16)a2.z; u.f[11] = (_Float16)a2.w;
  u.f[12] = (_Float16)a3.x; u.f[13] = (_Float16)a3.y;
  u.f[14] = (_Float16)a3.z; u.f[15] = (_Float16)a3.w;
  return u.v;
}
// B fragment gathered from a row-major [k][n] LDS tile (column access)
__device__ __forceinline__ v16h gather_bfrag(const _Float16* base, int rowStride,
                                             int col, int kbase, int half) {
  Frag16 u;
  const _Float16* p = base + (size_t)(kbase + half * 16) * rowStride + col;
#pragma unroll
  for (int e = 0; e < 16; ++e) u.f[e] = p[e * rowStride];
  return u.v;
}
// Pack float4 pair -> 8 halves
__device__ __forceinline__ v8h pack8(float4 a, float4 b) {
  v8h h;
  h[0] = (_Float16)a.x; h[1] = (_Float16)a.y; h[2] = (_Float16)a.z; h[3] = (_Float16)a.w;
  h[4] = (_Float16)b.x; h[5] = (_Float16)b.y; h[6] = (_Float16)b.z; h[7] = (_Float16)b.w;
  return h;
}

// ---------------------------------------------------------------------------
// K1: Y = X[8192x768] @ W[768x768] + bias, f16 out.
// layout 0: Q -> [s][bh][64] ; layout 1: K/V -> [bh][s][64]
// ---------------------------------------------------------------------------
__global__ void __launch_bounds__(256)
proj_kernel(const float* __restrict__ X, const float* __restrict__ W,
            const float* __restrict__ bias, _Float16* __restrict__ Y, int layout) {
  __shared__ __align__(16) _Float16 Xt[64][40];  // [m][c] f16
  __shared__ __align__(16) _Float16 Wt[64][40];  // transposed: [n][c]
  const int tid = threadIdx.x, lane = tid & 31, wave = tid >> 5;
  const int half = lane >> 4, l16 = lane & 15;
  const int wm = wave & 3, wn = wave >> 2;       // 4(m) x 2(n) waves
  const int m0 = blockIdx.x * 64;                // row R = b*S + s (x layout)
  const int n0 = blockIdx.y * 64;
  const int xi = tid * 8;                        // X-tile chunk: 8 halves/thread
  const int xr = xi >> 5, xc = xi & 31;
  const int wi = tid * 8;                        // W-tile chunk
  const int wr = wi >> 6, wc = wi & 63;

  v8f c0 = {}, c1 = {};
  for (int kk = 0; kk < WD; kk += 32) {
    __syncthreads();
    {   // stage X tile: 2x float4 -> one 16B LDS store
      const float4* src = (const float4*)(X + (size_t)(m0 + xr) * WD + kk + xc);
      *(v8h*)&Xt[xr][xc] = pack8(src[0], src[1]);
    }
    {   // stage W tile transposed: vector global read, 8 b16 scattered stores
      const float4* src = (const float4*)(W + (size_t)(kk + wr) * WD + n0 + wc);
      float4 w0 = src[0], w1 = src[1];
      Wt[wc + 0][wr] = (_Float16)w0.x; Wt[wc + 1][wr] = (_Float16)w0.y;
      Wt[wc + 2][wr] = (_Float16)w0.z; Wt[wc + 3][wr] = (_Float16)w0.w;
      Wt[wc + 4][wr] = (_Float16)w1.x; Wt[wc + 5][wr] = (_Float16)w1.y;
      Wt[wc + 6][wr] = (_Float16)w1.z; Wt[wc + 7][wr] = (_Float16)w1.w;
    }
    __syncthreads();
    v16h a  = load_afrag(&Xt[wm * 16 + l16][0], half);
    v16h b0 = load_bfrag(&Wt[wn * 32 + l16][0], half);
    v16h b1 = load_bfrag(&Wt[wn * 32 + 16 + l16][0], half);
    c0 = wmma_f16(a, b0, c0);
    c1 = wmma_f16(a, b1, c1);
  }
  FragC u[2]; u[0].v = c0; u[1].v = c1;
#pragma unroll
  for (int f = 0; f < 2; ++f) {
    int C = n0 + wn * 32 + f * 16 + l16;         // col 0..767 (= h*64+d)
    float bv = bias[C];
#pragma unroll
    for (int r = 0; r < 8; ++r) {
      int R = m0 + wm * 16 + 8 * half + r;
      int b = R >> 10, s = R & 1023;             // R = b*S + s
      float v = u[f].f[r] + bv;
      size_t addr = (layout == 0)
          ? ((size_t)(s * WB + b)) * WD + C
          : (((size_t)(b * WH + (C >> 6))) * WS + s) * 64 + (C & 63);
      Y[addr] = (_Float16)v;
    }
  }
}

// ---------------------------------------------------------------------------
// K2: per-j bias scores  SB[j][bh][k] = Q[j][bh][:] . pb[j][k][:]
// GEMM per j: [96 x 64] @ [64 x 1024]. 768 threads = 24 waves = 6(m) x 4(n).
// ---------------------------------------------------------------------------
__global__ void __launch_bounds__(768)
bias_score_kernel(const _Float16* __restrict__ Q, const float* __restrict__ PB,
                  _Float16* __restrict__ SB) {
  const int j = blockIdx.x;
  const int k0 = blockIdx.y * 64;
  const int tid = threadIdx.x, lane = tid & 31, wave = tid >> 5;
  const int half = lane >> 4, l16 = lane & 15;
  const int mi = wave % 6, ni = wave / 6;
  const int kcol = k0 + ni * 16 + l16;

  const _Float16* qrow = Q + ((size_t)j * WBH + mi * 16 + l16) * 64;
  v16h a0 = load_afrag(qrow, half);
  v16h a1 = load_afrag(qrow + 32, half);
  const float* prow = PB + ((size_t)j * WS + kcol) * 64;  // contiguous over d
  v16h b0 = load_bfrag_f32(prow, half);
  v16h b1 = load_bfrag_f32(prow + 32, half);

  v8f c = {};
  c = wmma_f16(a0, b0, c);
  c = wmma_f16(a1, b1, c);

  FragC u; u.v = c;
#pragma unroll
  for (int r = 0; r < 8; ++r) {
    int bh = mi * 16 + 8 * half + r;
    SB[((size_t)j * WBH + bh) * WS + kcol] = (_Float16)u.f[r];
  }
}

// ---------------------------------------------------------------------------
// K3: fused QK^T + bias + scale + v_mask + softmax -> A f16 [bh][j][k]
// One block per (bh, 16-row j-tile); 8 waves each own 1 k-tile per 128-k block.
// Scores kept in registers; row reductions via shuffles + small LDS.
// ---------------------------------------------------------------------------
__global__ void __launch_bounds__(256)
attn_score_softmax_kernel(const _Float16* __restrict__ Q, const _Float16* __restrict__ K,
                          const _Float16* __restrict__ SB, const float* __restrict__ vmask,
                          _Float16* __restrict__ A) {
  __shared__ __align__(16) _Float16 Kt[128][72];
  __shared__ float wmaxs[8][16];
  __shared__ float wsums[8][16];
  const int bh = blockIdx.x, j0 = blockIdx.y * 16;
  const int bb = bh / WH;
  const int tid = threadIdx.x, lane = tid & 31, wave = tid >> 5;
  const int half = lane >> 4, l16 = lane & 15;

  const _Float16* qrow = Q + ((size_t)(j0 + l16) * WBH + bh) * 64;
  v16h a0 = load_afrag(qrow, half);
  v16h a1 = load_afrag(qrow + 32, half);

  FragC s[8];
#pragma unroll
  for (int tb = 0; tb < 8; ++tb) {
    __syncthreads();
    {   // stage K[bh][tb*128 .. +127][0..63] -> Kt (half a row per thread)
      int row = tid >> 1, part = (tid & 1) * 32;
      const v8h* src = (const v8h*)(K + ((size_t)bh * WS + tb * 128 + row) * 64 + part);
      v8h* dst = (v8h*)&Kt[row][part];
      dst[0] = src[0]; dst[1] = src[1]; dst[2] = src[2]; dst[3] = src[3];
    }
    __syncthreads();
    int krow = wave * 16 + l16;                 // this lane's key (local)
    v16h b0 = load_bfrag(&Kt[krow][0], half);   // d 0..31
    v16h b1 = load_bfrag(&Kt[krow][32], half);  // d 32..63
    v8f c = {};
    c = wmma_f16(a0, b0, c);
    c = wmma_f16(a1, b1, c);
    int kg = tb * 128 + wave * 16 + l16;        // global k column
    float mk = (1.0f - vmask[bb * WS + kg]) * -1e12f;
    FragC u; u.v = c;
#pragma unroll
    for (int r = 0; r < 8; ++r) {
      int jg = j0 + 8 * half + r;
      float bias = (float)SB[((size_t)jg * WBH + bh) * WS + kg];
      u.f[r] = (u.f[r] + bias) * 0.125f + mk;
    }
    s[tb] = u;
  }

  // ---- softmax over k (1024) for the 16 rows ----
  float gm[8];
#pragma unroll
  for (int r = 0; r < 8; ++r) {
    float m = s[0].f[r];
#pragma unroll
    for (int tb = 1; tb < 8; ++tb) m = fmaxf(m, s[tb].f[r]);
#pragma unroll
    for (int d = 1; d < 16; d <<= 1) m = fmaxf(m, __shfl_xor(m, d, 16));
    gm[r] = m;
  }
  if (l16 == 0) {
#pragma unroll
    for (int r = 0; r < 8; ++r) wmaxs[wave][8 * half + r] = gm[r];
  }
  __syncthreads();
#pragma unroll
  for (int r = 0; r < 8; ++r) {
    float m = wmaxs[0][8 * half + r];
#pragma unroll
    for (int w = 1; w < 8; ++w) m = fmaxf(m, wmaxs[w][8 * half + r]);
    gm[r] = m;
  }
  float gs[8];
#pragma unroll
  for (int r = 0; r < 8; ++r) {
    float acc = 0.f;
#pragma unroll
    for (int tb = 0; tb < 8; ++tb) {
      float e = __expf(s[tb].f[r] - gm[r]);
      s[tb].f[r] = e;
      acc += e;
    }
#pragma unroll
    for (int d = 1; d < 16; d <<= 1) acc += __shfl_xor(acc, d, 16);
    gs[r] = acc;
  }
  if (l16 == 0) {
#pragma unroll
    for (int r = 0; r < 8; ++r) wsums[wave][8 * half + r] = gs[r];
  }
  __syncthreads();
#pragma unroll
  for (int r = 0; r < 8; ++r) {
    float t = 0.f;
#pragma unroll
    for (int w = 0; w < 8; ++w) t += wsums[w][8 * half + r];
    float inv = 1.0f / t;
    int jg = j0 + 8 * half + r;
#pragma unroll
    for (int tb = 0; tb < 8; ++tb) {
      int kg = tb * 128 + wave * 16 + l16;
      A[((size_t)bh * WS + jg) * WS + kg] = (_Float16)(s[tb].f[r] * inv);
    }
  }
}

// ---------------------------------------------------------------------------
// K4: O_main[j][b][h][d] = A[bh] @ V[bh]. Block = (bh, 16-row j-tile), 4 waves
// each own one 16-wide d slice and accumulate over k=1024.
// ---------------------------------------------------------------------------
__global__ void __launch_bounds__(128)
av_kernel(const _Float16* __restrict__ A, const _Float16* __restrict__ V,
          float* __restrict__ Om) {
  __shared__ __align__(16) _Float16 Vt[128][72];  // [k][d]
  const int bh = blockIdx.x, j0 = blockIdx.y * 16;
  const int b = bh / WH, h = bh % WH;
  const int tid = threadIdx.x, lane = tid & 31, wave = tid >> 5;
  const int half = lane >> 4, l16 = lane & 15;
  const int dcol = wave * 16 + l16;
  v8f c = {};
  for (int kb = 0; kb < 8; ++kb) {
    __syncthreads();
    {   // stage V: one 64-half row per thread
      const v8h* src = (const v8h*)(V + ((size_t)bh * WS + kb * 128 + tid) * 64);
      v8h* dst = (v8h*)&Vt[tid][0];
#pragma unroll
      for (int i = 0; i < 8; ++i) dst[i] = src[i];
    }
    if (kb < 7) __builtin_prefetch(V + ((size_t)bh * WS + (kb + 1) * 128 + tid) * 64, 0, 1);
    __syncthreads();
#pragma unroll
    for (int ks = 0; ks < 4; ++ks) {
      const _Float16* arow =
          A + ((size_t)bh * WS + j0 + l16) * WS + kb * 128 + ks * 32;
      v16h a  = load_afrag(arow, half);
      v16h bf = gather_bfrag(&Vt[0][0], 72, dcol, ks * 32, half);
      c = wmma_f16(a, bf, c);
    }
  }
  FragC u; u.v = c;
#pragma unroll
  for (int r = 0; r < 8; ++r) {
    int jg = j0 + 8 * half + r;
    Om[(((size_t)jg * WB + b) * WH + h) * 64 + dcol] = u.f[r];
  }
}

// ---------------------------------------------------------------------------
// K5: per-j output bias  Ob[j][b][h][d] = A[:, j, :] @ pb[j]
// One block per j: [96 x 1024] @ [1024 x 64]; 8 waves x 3 C-tiles (6m x 4n).
// ---------------------------------------------------------------------------
__global__ void __launch_bounds__(256)
abias_kernel(const _Float16* __restrict__ A, const float* __restrict__ PB,
             float* __restrict__ Ob) {
  __shared__ __align__(16) _Float16 Pt[64][72];  // [k][d] f16
  const int j = blockIdx.x;
  const int tid = threadIdx.x, lane = tid & 31, wave = tid >> 5;
  const int half = lane >> 4, l16 = lane & 15;
  v8f c[3] = {v8f{}, v8f{}, v8f{}};
  int mi[3], ni[3];
#pragma unroll
  for (int t = 0; t < 3; ++t) { int tt = wave + 8 * t; mi[t] = tt % 6; ni[t] = tt / 6; }

  for (int kb = 0; kb < 16; ++kb) {
    __syncthreads();
#pragma unroll
    for (int it = 0; it < 4; ++it) {    // stage pb tile: float4 -> 8B LDS store
      int i = (tid + it * 256) * 4;     // 4096 halves
      int kr = i >> 6, d0 = i & 63;
      float4 v = *(const float4*)(PB + ((size_t)j * WS + kb * 64 + kr) * 64 + d0);
      v4h h;
      h[0] = (_Float16)v.x; h[1] = (_Float16)v.y;
      h[2] = (_Float16)v.z; h[3] = (_Float16)v.w;
      *(v4h*)&Pt[kr][d0] = h;
    }
    __syncthreads();
#pragma unroll
    for (int ks = 0; ks < 2; ++ks) {
#pragma unroll
      for (int t = 0; t < 3; ++t) {
        const _Float16* arow =
            A + ((size_t)(mi[t] * 16 + l16) * WS + j) * WS + kb * 64 + ks * 32;
        v16h a  = load_afrag(arow, half);
        v16h bf = gather_bfrag(&Pt[0][0], 72, ni[t] * 16 + l16, ks * 32, half);
        c[t] = wmma_f16(a, bf, c[t]);
      }
    }
  }
#pragma unroll
  for (int t = 0; t < 3; ++t) {
    FragC u; u.v = c[t];
#pragma unroll
    for (int r = 0; r < 8; ++r) {
      int bh = mi[t] * 16 + 8 * half + r;
      int b = bh / WH, h = bh % WH;
      Ob[(((size_t)j * WB + b) * WH + h) * 64 + ni[t] * 16 + l16] = u.f[r];
    }
  }
}

// ---------------------------------------------------------------------------
// K6: out = ((Om+Ob)[8192x768] @ Wo + bo) * q_mask, f32 out [B,S,768].
// Input rows are R = s*B + b (memory order of Om/Ob).
// ---------------------------------------------------------------------------
__global__ void __launch_bounds__(256)
outproj_kernel(const float* __restrict__ Om, const float* __restrict__ Ob,
               const float* __restrict__ Wo, const float* __restrict__ bo,
               const float* __restrict__ qmask, float* __restrict__ out) {
  __shared__ __align__(16) _Float16 Xt[64][40];
  __shared__ __align__(16) _Float16 Wt[64][40];  // transposed [n][c]
  const int tid = threadIdx.x, lane = tid & 31, wave = tid >> 5;
  const int half = lane >> 4, l16 = lane & 15;
  const int wm = wave & 3, wn = wave >> 2;
  const int m0 = blockIdx.x * 64, n0 = blockIdx.y * 64;
  const int xi = tid * 8;
  const int xr = xi >> 5, xc = xi & 31;
  const int wi = tid * 8;
  const int wr = wi >> 6, wc = wi & 63;
  v8f c0 = {}, c1 = {};
  for (int kk = 0; kk < WD; kk += 32) {
    __syncthreads();
    {   // stage (Om+Ob) tile: 4x float4 -> one 16B LDS store
      size_t a = (size_t)(m0 + xr) * WD + kk + xc;
      const float4* pm = (const float4*)(Om + a);
      const float4* pb4 = (const float4*)(Ob + a);
      float4 m0v = pm[0], m1v = pm[1], b0v = pb4[0], b1v = pb4[1];
      float4 s0, s1;
      s0.x = m0v.x + b0v.x; s0.y = m0v.y + b0v.y; s0.z = m0v.z + b0v.z; s0.w = m0v.w + b0v.w;
      s1.x = m1v.x + b1v.x; s1.y = m1v.y + b1v.y; s1.z = m1v.z + b1v.z; s1.w = m1v.w + b1v.w;
      *(v8h*)&Xt[xr][xc] = pack8(s0, s1);
    }
    {   // stage Wo tile transposed
      const float4* src = (const float4*)(Wo + (size_t)(kk + wr) * WD + n0 + wc);
      float4 w0 = src[0], w1 = src[1];
      Wt[wc + 0][wr] = (_Float16)w0.x; Wt[wc + 1][wr] = (_Float16)w0.y;
      Wt[wc + 2][wr] = (_Float16)w0.z; Wt[wc + 3][wr] = (_Float16)w0.w;
      Wt[wc + 4][wr] = (_Float16)w1.x; Wt[wc + 5][wr] = (_Float16)w1.y;
      Wt[wc + 6][wr] = (_Float16)w1.z; Wt[wc + 7][wr] = (_Float16)w1.w;
    }
    __syncthreads();
    v16h a  = load_afrag(&Xt[wm * 16 + l16][0], half);
    v16h b0 = load_bfrag(&Wt[wn * 32 + l16][0], half);
    v16h b1 = load_bfrag(&Wt[wn * 32 + 16 + l16][0], half);
    c0 = wmma_f16(a, b0, c0);
    c1 = wmma_f16(a, b1, c1);
  }
  FragC u[2]; u[0].v = c0; u[1].v = c1;
#pragma unroll
  for (int f = 0; f < 2; ++f) {
    int C = n0 + wn * 32 + f * 16 + l16;
    float bv = bo[C];
#pragma unroll
    for (int r = 0; r < 8; ++r) {
      int R = m0 + wm * 16 + 8 * half + r;
      int s = R >> 3, b = R & 7;                 // R = s*B + b
      float qm = qmask[b * WS + s];
      out[((size_t)(b * WS + s)) * WD + C] = (u[f].f[r] + bv) * qm;
    }
  }
}

// ---------------------------------------------------------------------------
extern "C" void kernel_launch(void* const* d_in, const int* in_sizes, int n_in,
                              void* d_out, int out_size, void* d_ws, size_t ws_size,
                              hipStream_t stream) {
  (void)in_sizes; (void)n_in; (void)out_size; (void)ws_size;
  const float* x     = (const float*)d_in[0];
  const float* Wq    = (const float*)d_in[1];
  const float* bq    = (const float*)d_in[2];
  const float* Wk    = (const float*)d_in[3];
  const float* bk    = (const float*)d_in[4];
  const float* Wv    = (const float*)d_in[5];
  const float* bv    = (const float*)d_in[6];
  const float* Wo    = (const float*)d_in[7];
  const float* bo    = (const float*)d_in[8];
  const float* pb    = (const float*)d_in[9];
  const float* qmask = (const float*)d_in[10];
  const float* vmask = (const float*)d_in[11];
  float* out = (float*)d_out;

  // Workspace carving (~468 MB total; every buffer fully written before read).
  char* ws = (char*)d_ws;
  size_t off = 0;
  auto take = [&](size_t bytes) {
    char* p = ws + off;
    off = (off + bytes + 255) & ~(size_t)255;
    return p;
  };
  _Float16* Qh = (_Float16*)take((size_t)WS * WBH * 64 * 2);   // [s][bh][d]
  _Float16* Kh = (_Float16*)take((size_t)WBH * WS * 64 * 2);   // [bh][s][d]
  _Float16* Vh = (_Float16*)take((size_t)WBH * WS * 64 * 2);   // [bh][s][d]
  _Float16* SB = (_Float16*)take((size_t)WS * WBH * WS * 2);   // bias scores [j][bh][k]
  _Float16* Ah = (_Float16*)take((size_t)WBH * WS * WS * 2);   // probs [bh][j][k]
  float*    Om = (float*)take((size_t)WS * WB * WD * 4);       // [s][b][h*64+d]
  float*    Ob = (float*)take((size_t)WS * WB * WD * 4);       // [s][b][h*64+d]

  dim3 gP(128, 12);
  proj_kernel<<<gP, 256, 0, stream>>>(x, Wq, bq, Qh, 0);
  proj_kernel<<<gP, 256, 0, stream>>>(x, Wk, bk, Kh, 1);
  proj_kernel<<<gP, 256, 0, stream>>>(x, Wv, bv, Vh, 1);
  bias_score_kernel<<<dim3(WS, 16), 768, 0, stream>>>(Qh, pb, SB);
  attn_score_softmax_kernel<<<dim3(WBH, 64), 256, 0, stream>>>(Qh, Kh, SB, vmask, Ah);
  av_kernel<<<dim3(WBH, 64), 128, 0, stream>>>(Ah, Vh, Om);
  abias_kernel<<<dim3(WS), 256, 0, stream>>>(Ah, pb, Ob);
  outproj_kernel<<<dim3(128, 12), 256, 0, stream>>>(Om, Ob, Wo, bo, qmask, out);
}
